// AdaptiveConv2DMod_23021024706946
// MI455X (gfx1250) — matrix operand order repaired and checked by reading, not compiled, not run on hardware
//
#include <hip/hip_runtime.h>
#include <hip/hip_bf16.h>

typedef __attribute__((ext_vector_type(16))) __bf16          v16bf;
typedef __attribute__((ext_vector_type(8)))  float           v8f;
typedef __attribute__((ext_vector_type(16))) unsigned short  v16u;

#define B_    8
#define CIN   512
#define COUT  512
#define HW    64
#define HWP   66                     // x padded with zero halo: slot = x + 1, x in [-1..64]
#define NK    4
#define KTAPS 9
#define KTOT  (KTAPS * CIN)          // 4608, tap-major K for the weight operand
#define NPIX  (HW * HW)              // 4096

__device__ __forceinline__ unsigned short f2bf(float f) {
  unsigned int u = __builtin_bit_cast(unsigned int, f);
  unsigned int r = u + 0x7fffu + ((u >> 16) & 1u);
  return (unsigned short)(r >> 16);
}

// ---------------- kernel 0: mod = embed@mod_w^T + mod_b ; sel = softmax(embed@adapt_w^T + adapt_b)
__global__ __launch_bounds__(512) void k_mod_sel(const float* __restrict__ embed,
                                                 const float* __restrict__ mod_w,
                                                 const float* __restrict__ mod_b,
                                                 const float* __restrict__ adapt_w,
                                                 const float* __restrict__ adapt_b,
                                                 float* __restrict__ mod,
                                                 float* __restrict__ sel) {
  const int b = blockIdx.x;
  const int i = threadIdx.x;
  const float* e = embed + (size_t)b * CIN;
  float s = 0.f;
  for (int k = 0; k < CIN; ++k) s += e[k] * mod_w[(size_t)i * CIN + k];
  mod[(size_t)b * CIN + i] = s + mod_b[i];

  __shared__ float lg[NK];
  if (i < NK) {
    float t = 0.f;
    for (int k = 0; k < CIN; ++k) t += e[k] * adapt_w[(size_t)i * CIN + k];
    lg[i] = t + adapt_b[i];
  }
  __syncthreads();
  if (i == 0) {
    float m = lg[0];
    for (int n = 1; n < NK; ++n) m = fmaxf(m, lg[n]);
    float ex[NK], d = 0.f;
    for (int n = 0; n < NK; ++n) { ex[n] = __expf(lg[n] - m); d += ex[n]; }
    for (int n = 0; n < NK; ++n) sel[b * NK + n] = ex[n] / d;
  }
}

// ---------------- kernel 1: per-(b,o) combine/modulate/demodulate -> bf16 A in [b][o][tap*512+i]
__global__ __launch_bounds__(256) void k_weights(const float* __restrict__ weights,
                                                 const float* __restrict__ mod,
                                                 const float* __restrict__ sel,
                                                 unsigned short* __restrict__ wsA) {
  const int bo = blockIdx.x;
  const int b  = bo >> 9;
  const int o  = bo & 511;
  const int t  = threadIdx.x;

  const float s0 = sel[b * NK + 0], s1 = sel[b * NK + 1];
  const float s2 = sel[b * NK + 2], s3 = sel[b * NK + 3];
  const size_t nstride = (size_t)COUT * CIN * KTAPS;
  const size_t obase   = ((size_t)o * CIN) * KTAPS;

  float wloc[18];
  float ss = 0.f;
  #pragma unroll
  for (int u = 0; u < 18; ++u) {
    const int e = t + u * 256;            // 0..4607 over (i, tap)
    const int i = e / 9;
    const int r = e - i * 9;
    const size_t idx = obase + (size_t)i * 9 + r;
    float w = weights[idx] * s0 + weights[idx + nstride] * s1 +
              weights[idx + 2 * nstride] * s2 + weights[idx + 3 * nstride] * s3;
    w *= (mod[(size_t)b * CIN + i] + 1.0f);
    wloc[u] = w;
    ss += w * w;
  }
  for (int off = 16; off > 0; off >>= 1) ss += __shfl_down(ss, off, 32);
  __shared__ float red[8];
  if ((t & 31) == 0) red[t >> 5] = ss;
  __syncthreads();
  __shared__ float invn;
  if (t == 0) {
    float a = 0.f;
    for (int j = 0; j < 8; ++j) a += red[j];
    invn = rsqrtf(fmaxf(a, 1e-8f));
  }
  __syncthreads();
  const float iv = invn;

  unsigned short* dst = wsA + (size_t)(b * COUT + o) * KTOT;
  #pragma unroll
  for (int u = 0; u < 18; ++u) {
    const int e = t + u * 256;
    const int i = e / 9;
    const int r = e - i * 9;
    dst[(size_t)r * CIN + i] = f2bf(wloc[u] * iv);   // tap-major K
  }
}

// ---------------- kernel 2: fmap NCHW f32 -> x-padded NHWC bf16 (zero halo at x'=-1 and x'=64)
__global__ __launch_bounds__(256) void k_cvt_pad(const float* __restrict__ src,
                                                 unsigned short* __restrict__ dst) {
  const int cchunk = blockIdx.x;   // 0..15 (32 channels each)
  const int y      = blockIdx.y;   // 0..63
  const int b      = blockIdx.z;   // 0..7
  const int t = threadIdx.x;

  __shared__ unsigned short lds[64][40];   // [x][c], 80B rows (16B aligned)

  const int c  = t >> 3;            // 0..31
  const int xb = (t & 7) * 8;       // 0..56
  const float* sp = src + ((size_t)(b * CIN + cchunk * 32 + c) * HW + y) * HW + xb;
  #pragma unroll
  for (int j = 0; j < 8; ++j) lds[xb + j][c] = f2bf(sp[j]);
  __syncthreads();

  unsigned short* rowb = dst + (size_t)(b * HW + y) * HWP * CIN + cchunk * 32;

  const int x  = t >> 2;            // 0..63
  const int cb = (t & 3) * 8;       // 0..24
  *(uint4*)(rowb + (size_t)(x + 1) * CIN + cb) = *(const uint4*)&lds[x][cb];

  // zero halo columns (x slots 0 and 65) for this row / channel chunk
  if (t < 8) {
    const uint4 z = make_uint4(0u, 0u, 0u, 0u);
    const int slot = (t < 4) ? 0 : 65;
    *(uint4*)(rowb + (size_t)slot * CIN + (t & 3) * 8) = z;
  }
}

// ---------------- kernel 3: grouped conv as per-batch GEMM.
// Each wave owns a 64(M) x 64(N=1 image row) tile: 16 f32 accumulators.
// Branch-free inner loop: 8 A + 8 B global b128 loads -> 16 WMMAs per K-step.
__global__ __launch_bounds__(128) void k_conv(const unsigned short* __restrict__ wsA,
                                              const unsigned short* __restrict__ fpad,
                                              float* __restrict__ out) {
  const int ytile = blockIdx.x;   // 4 image rows per block
  const int mtile = blockIdx.y;   // 64 output channels
  const int b     = blockIdx.z;
  const int t     = threadIdx.x;
  const int lane  = t & 31;
  const int wave  = t >> 5;
  const int y     = ytile * 4 + wave;   // this wave's output row

  v8f acc[16] = {};

  const int col   = lane & 15;
  const int hi    = (lane >= 16) ? 1 : 0;
  const int koffA = hi ? 8 : 0;    // A 16-bit layout: lane half holds K{0..7,16..23}/{8..15,24..31}
  const int koffB = hi ? 8 : 0;    // B 32x16 layout: lane half holds K{0..7,16..23}/{8..15,24..31}

  const unsigned short* abase[4];
  #pragma unroll
  for (int m = 0; m < 4; ++m)
    abase[m] = wsA + (size_t)(b * COUT + mtile * 64 + m * 16 + col) * KTOT + koffA;

  for (int tap = 0; tap < KTAPS; ++tap) {
    const int ky = tap / 3, kx = tap - ky * 3;
    const int yy = y + ky - 1;
    if (yy < 0 || yy >= HW) continue;   // uniform per wave: zero rows are skipped, not masked

    const unsigned short* rowb = fpad + (size_t)(b * HW + yy) * HWP * CIN + koffB;
    const unsigned short* bp[4];
    #pragma unroll
    for (int ns = 0; ns < 4; ++ns)
      bp[ns] = rowb + (size_t)(ns * 16 + col + kx) * CIN;   // halo slot = x + 1
    const int kb0 = tap * CIN;

    for (int ic = 0; ic < CIN / 32; ++ic) {
      const int kc = ic * 32;

      // 4 A fragments (weights), reused across all 4 N-subtiles
      v16bf av[4];
      #pragma unroll
      for (int m = 0; m < 4; ++m) {
        union { uint4 q[2]; v16u v; } au;
        const unsigned short* ap = abase[m] + kb0 + kc;
        au.q[0] = *(const uint4*)ap;
        au.q[1] = *(const uint4*)(ap + 16);
        av[m] = __builtin_bit_cast(v16bf, au.v);
      }
      __builtin_prefetch((const void*)(abase[0] + kb0 + kc + 32), 0, 3);

      #pragma unroll
      for (int ns = 0; ns < 4; ++ns) {
        union { uint4 q[2]; v16u v; } bu;
        bu.q[0] = *(const uint4*)(bp[ns] + kc);
        bu.q[1] = *(const uint4*)(bp[ns] + kc + 16);
        const v16bf bv = __builtin_bit_cast(v16bf, bu.v);
        #pragma unroll
        for (int m = 0; m < 4; ++m) {
          acc[ns * 4 + m] = __builtin_amdgcn_wmma_f32_16x16x32_bf16(
              false, av[m], false, bv, (short)0, acc[ns * 4 + m], false, false);
        }
      }
    }
  }

  // D 16x16 f32 layout: lanes0-15 -> M=r, N=lane ; lanes16-31 -> M=r+8, N=lane-16
  const int Mofs = hi ? 8 : 0;
  #pragma unroll
  for (int ns = 0; ns < 4; ++ns) {
    #pragma unroll
    for (int m = 0; m < 4; ++m) {
      float* ob = out + (size_t)(b * COUT + mtile * 64 + m * 16) * NPIX + (size_t)y * HW;
      #pragma unroll
      for (int r = 0; r < 8; ++r)
        ob[(size_t)(r + Mofs) * NPIX + ns * 16 + col] = acc[ns * 4 + m][r];
    }
  }
}

extern "C" void kernel_launch(void* const* d_in, const int* in_sizes, int n_in,
                              void* d_out, int out_size, void* d_ws, size_t ws_size,
                              hipStream_t stream) {
  const float* fmap    = (const float*)d_in[0];
  const float* embed   = (const float*)d_in[1];
  const float* weights = (const float*)d_in[2];
  const float* mod_w   = (const float*)d_in[3];
  const float* mod_b   = (const float*)d_in[4];
  const float* adapt_w = (const float*)d_in[5];
  const float* adapt_b = (const float*)d_in[6];
  float* out = (float*)d_out;

  char* ws = (char*)d_ws;
  float* mod = (float*)(ws);                                   // 16 KB
  float* sel = (float*)(ws + 16384);                           // 128 B
  unsigned short* wsA  = (unsigned short*)(ws + 32768);        // 8*512*4608*2 = 36 MB
  unsigned short* fpad = (unsigned short*)(ws + 32768 + (size_t)B_ * COUT * KTOT * 2); // 8*64*66*512*2 = 34.6 MB

  hipLaunchKernelGGL(k_mod_sel, dim3(B_), dim3(512), 0, stream,
                     embed, mod_w, mod_b, adapt_w, adapt_b, mod, sel);
  hipLaunchKernelGGL(k_weights, dim3(B_ * COUT), dim3(256), 0, stream,
                     weights, mod, sel, wsA);
  hipLaunchKernelGGL(k_cvt_pad, dim3(CIN / 32, HW, B_), dim3(256), 0, stream,
                     fmap, fpad);
  hipLaunchKernelGGL(k_conv, dim3(HW / 4, COUT / 64, B_), dim3(128), 0, stream,
                     wsA, fpad, out);
}
